// EfficientMoE_1769526526606
// MI455X (gfx1250) — matrix operand order repaired and checked by reading, not compiled, not run on hardware
//
#include <hip/hip_runtime.h>
#include <hip/hip_bf16.h>

// MoE: D_MODEL=1024, D_FF=4096, E=8, TOP_K=2, T=B*S=8192
#define DM 1024
#define FF 4096
#define NE 8
#define TT 8192
#define TM 32   // tokens per workgroup tile

typedef __attribute__((ext_vector_type(16))) __bf16 v16bf;
typedef __attribute__((ext_vector_type(8)))  float  v8f;

union BF16x16 { v16bf v; uint4 q[2]; };

static __device__ __forceinline__ unsigned short f2bf(float f) {
  unsigned int u = __float_as_uint(f);
  unsigned int r = (u + 0x7FFFu + ((u >> 16) & 1u)) >> 16;   // RNE
  return (unsigned short)r;
}

// ---------------- transpose + f32->bf16 convert:  src[e][r][c] -> dst[e][c][r]
__global__ void moe_transpose_bf16(const float* __restrict__ src,
                                   unsigned short* __restrict__ dst,
                                   int R, int C) {
  long long idx = (long long)blockIdx.x * 256 + threadIdx.x;
  long long total = (long long)NE * R * C;
  if (idx >= total) return;
  int e = (int)(idx / ((long long)R * C));
  long long rem = idx - (long long)e * R * C;
  int c = (int)(rem / R);
  int r = (int)(rem % R);
  dst[idx] = f2bf(src[((long long)e * R + r) * C + c]);
}

// ---------------- router: one wave per token ----------------
__global__ void moe_router(const float* __restrict__ x,
                           const float* __restrict__ Wg,
                           const float* __restrict__ bg,
                           float* __restrict__ wdense,
                           float* __restrict__ probsum) {
  int wave = threadIdx.x >> 5;
  int lane = threadIdx.x & 31;
  int t = blockIdx.x * 8 + wave;
  if (t >= TT) return;

  float s[NE];
#pragma unroll
  for (int e = 0; e < NE; ++e) s[e] = 0.f;

  const float* xr = x + (size_t)t * DM;
  for (int k = lane; k < DM; k += 32) {
    float xv = xr[k];
    const float* wr = Wg + (size_t)k * NE;
#pragma unroll
    for (int e = 0; e < NE; ++e) s[e] += xv * wr[e];
  }
#pragma unroll
  for (int off = 16; off > 0; off >>= 1) {
#pragma unroll
    for (int e = 0; e < NE; ++e) s[e] += __shfl_xor(s[e], off, 32);
  }
#pragma unroll
  for (int e = 0; e < NE; ++e) s[e] += bg[e];

  float mx = s[0];
#pragma unroll
  for (int e = 1; e < NE; ++e) mx = fmaxf(mx, s[e]);
  float p[NE], den = 0.f;
#pragma unroll
  for (int e = 0; e < NE; ++e) { p[e] = __expf(s[e] - mx); den += p[e]; }
  float inv = 1.f / den;
#pragma unroll
  for (int e = 0; e < NE; ++e) p[e] *= inv;

  // top-2, first-index tie break (matches jax.lax.top_k)
  int i1 = 0;
#pragma unroll
  for (int e = 1; e < NE; ++e) if (p[e] > p[i1]) i1 = e;
  int i2 = (i1 == 0) ? 1 : 0;
#pragma unroll
  for (int e = 0; e < NE; ++e) if (e != i1 && p[e] > p[i2]) i2 = e;
  float sum2 = p[i1] + p[i2];
  float g1 = p[i1] / sum2, g2 = p[i2] / sum2;

  if (lane < NE) {
    float wv = (lane == i1) ? g1 : ((lane == i2) ? g2 : 0.f);
    wdense[(size_t)t * NE + lane] = wv;
    atomicAdd(&probsum[lane], p[lane]);
  }
}

// ---------------- load-balance loss (unbiased variance over expert means) ----
__global__ void moe_loss(const float* __restrict__ probsum, float* __restrict__ out_loss) {
  if (threadIdx.x == 0 && blockIdx.x == 0) {
    float m[NE], avg = 0.f;
#pragma unroll
    for (int e = 0; e < NE; ++e) { m[e] = probsum[e] / (float)TT; avg += m[e]; }
    avg *= (1.f / NE);
    float v = 0.f;
#pragma unroll
    for (int e = 0; e < NE; ++e) { float d = m[e] - avg; v += d * d; }
    out_loss[0] = v / (float)(NE - 1);
  }
}

// ---------------- fused expert FFN with bf16 WMMA ----------------
// grid.x = T/TM token tiles; block = 256 (8 waves). All experts looped inside.
// Gate weight is folded into the bf16 h-chunk, so GEMM2 accumulates the final
// gated output across all experts directly in registers (no global RMW).
__global__ __launch_bounds__(256) void moe_ffn(
    const float* __restrict__ x, const float* __restrict__ b1,
    const float* __restrict__ b2, const unsigned short* __restrict__ W1T,
    const unsigned short* __restrict__ W2T, const float* __restrict__ wdense,
    float* __restrict__ out) {
  __shared__ __align__(16) unsigned short Xs[TM * DM];   // 64 KB bf16 token tile
  __shared__ __align__(16) unsigned short Hs[TM * 256];  // 16 KB bf16 h-chunk
  __shared__ float Ws[TM * NE];                          // 1 KB gate weights

  const int tid = threadIdx.x;
  const int w = tid >> 5;          // wave 0..7
  const int lane = tid & 31;
  const int r  = lane & 15;        // A-matrix row within 16-tile
  const int hi = lane >> 4;        // lane-half
  const int nl = lane & 15;        // B/C column within 16-tile
  const int t0 = blockIdx.x * TM;

  // load + convert the TM x 1024 token tile (contiguous -> coalesced float4)
  const float4* xsrc = reinterpret_cast<const float4*>(x) + (size_t)t0 * (DM / 4);
  for (int i = tid; i < TM * DM / 4; i += 256) {
    float4 v = xsrc[i];
    ushort4 b;
    b.x = f2bf(v.x); b.y = f2bf(v.y); b.z = f2bf(v.z); b.w = f2bf(v.w);
    reinterpret_cast<ushort4*>(Xs)[i] = b;
  }
  // stage the TM x NE gate weights (exactly 256 floats)
  Ws[tid] = wdense[(size_t)t0 * NE + tid];
  __syncthreads();

  const v8f vzero = {0.f, 0.f, 0.f, 0.f, 0.f, 0.f, 0.f, 0.f};

  // final gated output accumulators: 2 M-tiles x 8 N-tiles = 128 VGPRs
  v8f facc[2][8];
#pragma unroll
  for (int m = 0; m < 2; ++m)
#pragma unroll
    for (int n = 0; n < 8; ++n) facc[m][n] = vzero;

  for (int e = 0; e < NE; ++e) {
    for (int c = 0; c < FF / 256; ++c) {
      const int f0 = c * 256;
      const int fw = f0 + w * 32;           // this wave's 32 h-columns

      // ---- Phase A: h[TM x 32] = relu(X @ W1_e[:, fw:fw+32] + b1)
      v8f ha[2][2];
#pragma unroll
      for (int m = 0; m < 2; ++m) { ha[m][0] = vzero; ha[m][1] = vzero; }

      for (int kb = 0; kb < DM; kb += 32) {
        BF16x16 a0, a1;
        a0.q[0] = *reinterpret_cast<const uint4*>(&Xs[r * DM + kb + 8 * hi]);
        a0.q[1] = *reinterpret_cast<const uint4*>(&Xs[r * DM + kb + 16 + 8 * hi]);
        a1.q[0] = *reinterpret_cast<const uint4*>(&Xs[(16 + r) * DM + kb + 8 * hi]);
        a1.q[1] = *reinterpret_cast<const uint4*>(&Xs[(16 + r) * DM + kb + 16 + 8 * hi]);
#pragma unroll
        for (int n = 0; n < 2; ++n) {
          const unsigned short* bp =
              W1T + ((size_t)(e * FF + fw + n * 16 + nl)) * DM + kb + 8 * hi;
          __builtin_prefetch(bp + 32, 0, 1);   // next k-slab
          BF16x16 b;
          b.q[0] = *reinterpret_cast<const uint4*>(bp);
          b.q[1] = *reinterpret_cast<const uint4*>(bp + 16);
          ha[0][n] = __builtin_amdgcn_wmma_f32_16x16x32_bf16(false, a0.v, false, b.v,
                                                             (short)0, ha[0][n], false, false);
          ha[1][n] = __builtin_amdgcn_wmma_f32_16x16x32_bf16(false, a1.v, false, b.v,
                                                             (short)0, ha[1][n], false, false);
        }
      }
      __syncthreads();   // prior Phase-B readers of Hs are done
#pragma unroll
      for (int n = 0; n < 2; ++n) {
        float bias = b1[e * FF + fw + n * 16 + nl];
#pragma unroll
        for (int m = 0; m < 2; ++m) {
#pragma unroll
          for (int i = 0; i < 8; ++i) {
            int tl = m * 16 + i + 8 * hi;              // local token row
            float wt = Ws[tl * NE + e];                // gate weight
            float hv = ha[m][n][i] + bias;
            hv = hv > 0.f ? hv : 0.f;
            Hs[tl * 256 + (w * 32 + n * 16 + nl)] = f2bf(hv * wt);
          }
        }
      }
      __syncthreads();   // Hs chunk visible

      // ---- Phase B: facc += (wt*h_chunk) @ W2_e[f0:f0+256, :]
      for (int kb = 0; kb < 256; kb += 32) {
        BF16x16 a0, a1;
        a0.q[0] = *reinterpret_cast<const uint4*>(&Hs[r * 256 + kb + 8 * hi]);
        a0.q[1] = *reinterpret_cast<const uint4*>(&Hs[r * 256 + kb + 16 + 8 * hi]);
        a1.q[0] = *reinterpret_cast<const uint4*>(&Hs[(16 + r) * 256 + kb + 8 * hi]);
        a1.q[1] = *reinterpret_cast<const uint4*>(&Hs[(16 + r) * 256 + kb + 16 + 8 * hi]);
#pragma unroll
        for (int n = 0; n < 8; ++n) {
          const unsigned short* bp =
              W2T + ((size_t)(e * DM + w * 128 + n * 16 + nl)) * FF + f0 + kb + 8 * hi;
          BF16x16 b;
          b.q[0] = *reinterpret_cast<const uint4*>(bp);
          b.q[1] = *reinterpret_cast<const uint4*>(bp + 16);
          facc[0][n] = __builtin_amdgcn_wmma_f32_16x16x32_bf16(false, a0.v, false, b.v,
                                                               (short)0, facc[0][n], false, false);
          facc[1][n] = __builtin_amdgcn_wmma_f32_16x16x32_bf16(false, a1.v, false, b.v,
                                                               (short)0, facc[1][n], false, false);
        }
      }
    }
  }

  // ---- final epilogue: out = facc + sum_e wt[t,e]*b2[e,d]  (single store pass)
#pragma unroll
  for (int n = 0; n < 8; ++n) {
    int d = w * 128 + n * 16 + nl;
    float bb[NE];
#pragma unroll
    for (int e = 0; e < NE; ++e) bb[e] = b2[e * DM + d];
#pragma unroll
    for (int m = 0; m < 2; ++m) {
#pragma unroll
      for (int i = 0; i < 8; ++i) {
        int tl = m * 16 + i + 8 * hi;
        float acc = facc[m][n][i];
#pragma unroll
        for (int e = 0; e < NE; ++e) acc += Ws[tl * NE + e] * bb[e];
        out[(size_t)(t0 + tl) * DM + d] = acc;
      }
    }
  }
}

extern "C" void kernel_launch(void* const* d_in, const int* in_sizes, int n_in,
                              void* d_out, int out_size, void* d_ws, size_t ws_size,
                              hipStream_t stream) {
  (void)in_sizes; (void)n_in; (void)out_size; (void)ws_size;
  const float* x  = (const float*)d_in[0];
  const float* Wg = (const float*)d_in[1];
  const float* bg = (const float*)d_in[2];
  const float* W1 = (const float*)d_in[3];
  const float* b1 = (const float*)d_in[4];
  const float* W2 = (const float*)d_in[5];
  const float* b2 = (const float*)d_in[6];
  float* out = (float*)d_out;

  // workspace layout (~134.5 MB total):
  //   [0,32)                probsum f32[8]
  //   [1024, +256KB)        dense gate weights f32[T][8]
  //   then W1T bf16 [8][4096][1024]  (64 MB)
  //   then W2T bf16 [8][1024][4096]  (64 MB)
  char* ws = (char*)d_ws;
  float* probsum = (float*)ws;
  float* wdense  = (float*)(ws + 1024);
  unsigned short* W1T = (unsigned short*)(ws + 1024 + (size_t)TT * NE * 4);
  unsigned short* W2T = W1T + (size_t)NE * FF * DM;

  hipMemsetAsync(probsum, 0, 64, stream);

  {
    long long total = (long long)NE * DM * FF;
    int blocks = (int)((total + 255) / 256);
    moe_transpose_bf16<<<blocks, 256, 0, stream>>>(W1, W1T, DM, FF);  // -> [e][f][d]
    moe_transpose_bf16<<<blocks, 256, 0, stream>>>(W2, W2T, FF, DM);  // -> [e][d][f]
  }

  moe_router<<<TT / 8, 256, 0, stream>>>(x, Wg, bg, wdense, probsum);
  moe_loss<<<1, 32, 0, stream>>>(probsum, out + (size_t)TT * DM);
  moe_ffn<<<TT / TM, 256, 0, stream>>>(x, b1, b2, W1T, W2T, wdense, out);
}
